// RM3ExpertPool_24653112279097
// MI455X (gfx1250) — matrix operand order repaired and checked by reading, not compiled, not run on hardware
//
#include <hip/hip_runtime.h>
#include <hip/hip_bf16.h>
#include <math.h>

// ---------------------------------------------------------------------------
// RM3ExpertPool for MI455X (gfx1250): bf16 WMMA pipeline.
//   K1: err[e,b]  = || x - x@W_align[e] ||^2    (fused norm epilogue, no x_al)
//   K2: winner[b] = argmin_e err[e,b]           (== argmax exp(-sqrt(err)))
//   K3: [gate|xval|dt|phase] = x @ Mt[e], Mt = composite(W_align @ W_cat^T),
//       elementwise recurrent update -> NR[e,b,r] masked by winner
//   K4: out = sum_e NR[e] @ W_out[e]^T          (masking == one-hot select)
// GEMMs: V_WMMA_F32_16X16X32_BF16; wave computes a (MT*16) x 64 strip so each
// B fragment is reused MT times (MT=4 for the dominant GEMM -> ~32 FLOP/B of
// L2 traffic instead of ~13).
// ---------------------------------------------------------------------------

typedef __attribute__((ext_vector_type(16))) __bf16 v16bf;
typedef __attribute__((ext_vector_type(8)))  __bf16 v8bf;
typedef __attribute__((ext_vector_type(8)))  float  v8f;

#define NB 16384   // batch rows
#define ND 1024    // feature dim
#define RD 128     // recurrent dim
#define NE 8       // experts
#define NCAT 512   // gate(128)|xval(128)|dt(128)|phase(128)
#define PI_F 3.14159265358979323846f

// ---- WMMA fragment loaders (ISA 7.12.2 layouts) ---------------------------
// A (16x32 bf16, row-major source): lane l -> m = l&15, h = l>>4.
// element i<8  : K = h*8 + i ; element i>=8 : K = 16 + h*8 + (i-8).
__device__ __forceinline__ v16bf load_a_frag(const __bf16* __restrict__ A,
                                             int lda, int m0, int k0, int lane) {
  int m = m0 + (lane & 15);
  int h = lane >> 4;
  const __bf16* p = A + (size_t)m * lda + (size_t)(k0 + h * 8);
  v8bf lo = *reinterpret_cast<const v8bf*>(p);
  v8bf hi = *reinterpret_cast<const v8bf*>(p + 16);
  return __builtin_shufflevector(lo, hi, 0,1,2,3,4,5,6,7,8,9,10,11,12,13,14,15);
}

// B (32x16 bf16) from (N,K)-row-major weights: lane l -> n = l&15, h = l>>4.
// element i<8  : K = h*8 + i ; element i>=8 : K = 16 + h*8 + (i-8).
__device__ __forceinline__ v16bf load_b_frag(const __bf16* __restrict__ Wt,
                                             int ldb, int n0, int k0, int lane) {
  int n = n0 + (lane & 15);
  int h = lane >> 4;
  const __bf16* p = Wt + (size_t)n * ldb + (size_t)(k0 + h * 8);
  v8bf lo = *reinterpret_cast<const v8bf*>(p);
  v8bf hi = *reinterpret_cast<const v8bf*>(p + 16);
  return __builtin_shufflevector(lo, hi, 0,1,2,3,4,5,6,7,8,9,10,11,12,13,14,15);
}

// Wave-level (MT*16)(M) x 64(N) strip, accumulate into acc[MT][4].
// C/D tile layout: value acc[mm][t][v] is row (mm*16 + h*8 + v), col (n0+t*16+n).
template <int MT>
__device__ __forceinline__ void wave_gemm(const __bf16* __restrict__ A, int lda, int m0,
                                          const __bf16* __restrict__ Bw, int ldb, int n0,
                                          int K, int lane, v8f (&acc)[MT][4]) {
  for (int k0 = 0; k0 < K; k0 += 32) {
    v16bf a[MT];
#pragma unroll
    for (int mm = 0; mm < MT; ++mm)
      a[mm] = load_a_frag(A, lda, m0 + 16 * mm, k0, lane);
#pragma unroll
    for (int t = 0; t < 4; ++t) {
      v16bf b = load_b_frag(Bw, ldb, n0 + 16 * t, k0, lane);
#pragma unroll
      for (int mm = 0; mm < MT; ++mm)
        acc[mm][t] = __builtin_amdgcn_wmma_f32_16x16x32_bf16(
            false, a[mm], false, b, (short)0, acc[mm][t], false, false);
    }
  }
}

// ---- Pre-pass: conversions / layout builds --------------------------------
__global__ void k_cvt_f32_bf16(const float* __restrict__ src, __bf16* __restrict__ dst, size_t n) {
  size_t i = (size_t)blockIdx.x * blockDim.x + threadIdx.x;
  size_t stride = (size_t)gridDim.x * blockDim.x;
  for (; i < n; i += stride) dst[i] = (__bf16)src[i];
}

// W_align (E,D,D)(d,f) -> WaT (E,f,d) bf16  (B-operand layout for err GEMM)
__global__ void k_cvt_alignT(const float* __restrict__ W, __bf16* __restrict__ WaT) {
  size_t n = (size_t)NE * ND * ND;
  size_t i = (size_t)blockIdx.x * blockDim.x + threadIdx.x;
  size_t stride = (size_t)gridDim.x * blockDim.x;
  for (; i < n; i += stride) {
    size_t e = i >> 20;            // D*D = 1<<20
    size_t rem = i & ((1u << 20) - 1);
    size_t d = rem >> 10, f = rem & 1023;
    WaT[(e << 20) + (f << 10) + d] = (__bf16)W[i];
  }
}

// Wcat rows: [0,128)=W_in gate rows, [128,256)=W_in xval rows, [256,384)=W_dt, [384,512)=W_phase
__global__ void k_build_wcat(const float* __restrict__ Win, const float* __restrict__ Wdt,
                             const float* __restrict__ Wph, __bf16* __restrict__ Wcat) {
  size_t n = (size_t)NE * NCAT * ND;
  size_t i = (size_t)blockIdx.x * blockDim.x + threadIdx.x;
  size_t stride = (size_t)gridDim.x * blockDim.x;
  for (; i < n; i += stride) {
    size_t e = i / ((size_t)NCAT * ND);
    size_t rem = i % ((size_t)NCAT * ND);
    size_t o = rem >> 10, f = rem & 1023;
    float v;
    if (o < 256)      v = Win[e * (size_t)(4 * RD) * ND + o * ND + f];
    else if (o < 384) v = Wdt[e * (size_t)RD * ND + (o - 256) * ND + f];
    else              v = Wph[e * (size_t)RD * ND + (o - 384) * ND + f];
    Wcat[i] = (__bf16)v;
  }
}

// Composite: Mt[e][o][d] = sum_f Wcat[e][o][f] * W_align_bf16[e][d][f]
__global__ void k_gemm_comp(const __bf16* __restrict__ Wcat, const __bf16* __restrict__ Wa,
                            __bf16* __restrict__ Mt) {
  int e = blockIdx.y;
  int rt = blockIdx.x >> 1;            // 32 row tiles of 16 (M = 512)
  int cb = blockIdx.x & 1;             // 2 col halves of 512 (N = 1024)
  int lane = threadIdx.x & 31, wave = threadIdx.x >> 5;
  const __bf16* Ae = Wcat + ((size_t)e << 19);
  const __bf16* Be = Wa + ((size_t)e << 20);
  __bf16* Ce = Mt + ((size_t)e << 19);
  int m0 = rt * 16;
  int n0 = cb * 512 + wave * 64;
  v8f acc[1][4] = {{v8f(0.f), v8f(0.f), v8f(0.f), v8f(0.f)}};
  wave_gemm<1>(Ae, ND, m0, Be, ND, n0, ND, lane, acc);
  int h = lane >> 4, n = lane & 15;
#pragma unroll
  for (int t = 0; t < 4; ++t)
#pragma unroll
    for (int v = 0; v < 8; ++v)
      Ce[(size_t)(m0 + h * 8 + v) * ND + (n0 + t * 16 + n)] = (__bf16)acc[0][t][v];
}

__global__ void k_zero_f32(float* __restrict__ p, size_t n) {
  size_t i = (size_t)blockIdx.x * blockDim.x + threadIdx.x;
  if (i < n) p[i] = 0.f;
}

// ---- K1: err[e,b] = sum_f (x - x@W_align[e])^2, deterministic (no atomics) --
// Block: 64 rows x 1024 cols (8 waves x 64 cols, 2 col passes), MT=4.
__global__ void k_err(const __bf16* __restrict__ xbf, const __bf16* __restrict__ WaT,
                      const float* __restrict__ x, float* __restrict__ err) {
  __shared__ float sred[8][64];        // [wave][row]
  int e = blockIdx.y;
  int rt = blockIdx.x;
  int lane = threadIdx.x & 31, wave = threadIdx.x >> 5;
  int m0 = rt * 64;
  int h = lane >> 4, n = lane & 15;
  const __bf16* Be = WaT + ((size_t)e << 20);
  float rs[4][8];
#pragma unroll
  for (int mm = 0; mm < 4; ++mm)
#pragma unroll
    for (int v = 0; v < 8; ++v) rs[mm][v] = 0.f;
  for (int cb = 0; cb < 2; ++cb) {     // 2 x 512 columns of f
    int n0 = cb * 512 + wave * 64;
    v8f acc[4][4];
#pragma unroll
    for (int mm = 0; mm < 4; ++mm)
#pragma unroll
      for (int t = 0; t < 4; ++t) acc[mm][t] = v8f(0.f);
    wave_gemm<4>(xbf, ND, m0, Be, ND, n0, ND, lane, acc);
#pragma unroll
    for (int mm = 0; mm < 4; ++mm)
#pragma unroll
      for (int t = 0; t < 4; ++t)
#pragma unroll
        for (int v = 0; v < 8; ++v) {
          float d = acc[mm][t][v] -
                    x[(size_t)(m0 + mm * 16 + h * 8 + v) * ND + (n0 + t * 16 + n)];
          rs[mm][v] += d * d;
        }
  }
  // reduce across the 16 lanes of each half (row = mm*16 + h*8 + v)
#pragma unroll
  for (int off = 1; off < 16; off <<= 1)
#pragma unroll
    for (int mm = 0; mm < 4; ++mm)
#pragma unroll
      for (int v = 0; v < 8; ++v) rs[mm][v] += __shfl_xor(rs[mm][v], off, 32);
  if (n == 0)
#pragma unroll
    for (int mm = 0; mm < 4; ++mm)
#pragma unroll
      for (int v = 0; v < 8; ++v) sred[wave][mm * 16 + h * 8 + v] = rs[mm][v];
  __syncthreads();
  if (threadIdx.x < 64) {
    float s = 0.f;
#pragma unroll
    for (int w = 0; w < 8; ++w) s += sred[w][threadIdx.x];
    err[(size_t)e * NB + m0 + threadIdx.x] = s;
  }
}

// ---- K2: winner = argmin_e err (first-min tie-break == jnp.argmax of fid) --
__global__ void k_winner(const float* __restrict__ err, int* __restrict__ winner) {
  int b = blockIdx.x * blockDim.x + threadIdx.x;
  if (b >= NB) return;
  float best = err[b];
  int bi = 0;
  for (int e = 1; e < NE; ++e) {
    float v = err[(size_t)e * NB + b];
    if (v < best) { best = v; bi = e; }
  }
  winner[b] = bi;
}

// ---- K3: proj = x@Mt[e] -> recurrent update -> NR masked by winner ---------
// Block: 32 rows x 512 cols (8 waves x 64 cols), MT=2; dynamic LDS 64 KB.
__global__ void k_cat(const __bf16* __restrict__ xbf, const __bf16* __restrict__ Mt,
                      const float* __restrict__ A_log, const float* __restrict__ A_imag,
                      const float* __restrict__ state_re, const float* __restrict__ state_im,
                      const int* __restrict__ winner, __bf16* __restrict__ NRm) {
  extern __shared__ float lp[];        // [32][NCAT] = 64 KB of the 320 KB LDS
  int e = blockIdx.y;
  int rt = blockIdx.x;
  int lane = threadIdx.x & 31, wave = threadIdx.x >> 5;
  int m0 = rt * 32;
  int n0 = wave * 64;
  v8f acc[2][4];
#pragma unroll
  for (int mm = 0; mm < 2; ++mm)
#pragma unroll
    for (int t = 0; t < 4; ++t) acc[mm][t] = v8f(0.f);
  wave_gemm<2>(xbf, ND, m0, Mt + ((size_t)e << 19), ND, n0, ND, lane, acc);
  int h = lane >> 4, n = lane & 15;
#pragma unroll
  for (int mm = 0; mm < 2; ++mm)
#pragma unroll
    for (int t = 0; t < 4; ++t)
#pragma unroll
      for (int v = 0; v < 8; ++v)
        lp[(mm * 16 + h * 8 + v) * NCAT + (n0 + t * 16 + n)] = acc[mm][t][v];
  __syncthreads();
  for (int i = threadIdx.x; i < 32 * RD; i += 256) {
    int row = i >> 7, r = i & (RD - 1);
    int b = m0 + row;
    float gate = lp[row * NCAT + r];
    float xval = lp[row * NCAT + RD + r];
    float dtp  = lp[row * NCAT + 2 * RD + r];
    float php  = lp[row * NCAT + 3 * RD + r];
    float dt = fmaxf(dtp, 0.f) + log1pf(__expf(-fabsf(dtp)));   // softplus
    float ph = tanhf(php) * PI_F;
    float al = A_log[e * RD + r], ai = A_imag[e * RD + r];
    float decay = __expf(-dt * __expf(al));
    float ang = dt * ai + ph;
    float c = __cosf(ang), s = __sinf(ang);
    size_t si = ((size_t)e * NB + b) * RD + r;
    float sig = 1.f / (1.f + __expf(-gate));
    float nre = (state_re[si] * c - state_im[si] * s) * decay + sig * xval;
    NRm[si] = (__bf16)((winner[b] == e) ? nre : 0.f);
  }
}

// ---- K4: out = sum_e NR[e] @ W_out[e]^T  (masked NR == one-hot select) -----
// Block: 32 rows x 512 cols (8 waves x 64 cols), MT=2.
__global__ void k_out(const __bf16* __restrict__ NRm, const __bf16* __restrict__ WoT,
                      float* __restrict__ out) {
  int rt = blockIdx.x >> 1;
  int cb = blockIdx.x & 1;
  int lane = threadIdx.x & 31, wave = threadIdx.x >> 5;
  int m0 = rt * 32;
  int n0 = cb * 512 + wave * 64;
  v8f acc[2][4];
#pragma unroll
  for (int mm = 0; mm < 2; ++mm)
#pragma unroll
    for (int t = 0; t < 4; ++t) acc[mm][t] = v8f(0.f);
  for (int e = 0; e < NE; ++e)
    wave_gemm<2>(NRm + ((size_t)e << 21), RD, m0,
                 WoT + ((size_t)e << 17), RD, n0, RD, lane, acc);
  int h = lane >> 4, n = lane & 15;
#pragma unroll
  for (int mm = 0; mm < 2; ++mm)
#pragma unroll
    for (int t = 0; t < 4; ++t)
#pragma unroll
      for (int v = 0; v < 8; ++v)
        out[(size_t)(m0 + mm * 16 + h * 8 + v) * ND + (n0 + t * 16 + n)] = acc[mm][t][v];
}

// ---------------------------------------------------------------------------
extern "C" void kernel_launch(void* const* d_in, const int* in_sizes, int n_in,
                              void* d_out, int out_size, void* d_ws, size_t ws_size,
                              hipStream_t stream) {
  (void)in_sizes; (void)n_in; (void)out_size; (void)ws_size;
  const float* x        = (const float*)d_in[0];   // (B,D)
  const float* W_align  = (const float*)d_in[1];   // (E,D,D)
  const float* W_in     = (const float*)d_in[2];   // (E,4R,D)
  const float* W_dt     = (const float*)d_in[3];   // (E,R,D)
  const float* W_phase  = (const float*)d_in[4];   // (E,R,D)
  const float* W_out    = (const float*)d_in[5];   // (E,D,R)
  const float* A_log    = (const float*)d_in[6];   // (E,R)
  const float* A_imag   = (const float*)d_in[7];   // (E,R)
  const float* state_re = (const float*)d_in[8];   // (E,B,R)
  const float* state_im = (const float*)d_in[9];   // (E,B,R)
  float* out = (float*)d_out;                      // (B,D)

  // Workspace carve-up (~120 MB), 256B aligned regions.
  char* base = (char*)d_ws;
  size_t off = 0;
  auto carve = [&](size_t bytes) { char* p = base + off; off += (bytes + 255) & ~(size_t)255; return p; };
  __bf16* xbf  = (__bf16*)carve((size_t)NB * ND * 2);           // x in bf16
  __bf16* WaT  = (__bf16*)carve((size_t)NE * ND * ND * 2);      // W_align^T bf16
  __bf16* Wa   = (__bf16*)carve((size_t)NE * ND * ND * 2);      // W_align bf16 (row-major)
  __bf16* Wcat = (__bf16*)carve((size_t)NE * NCAT * ND * 2);    // [gate|xval|dt|phase] rows
  __bf16* Mt   = (__bf16*)carve((size_t)NE * NCAT * ND * 2);    // composite weights
  __bf16* WoT  = (__bf16*)carve((size_t)NE * ND * RD * 2);      // W_out bf16 (already (N,K))
  __bf16* NRm  = (__bf16*)carve((size_t)NE * NB * RD * 2);      // masked new_re
  float*  err  = (float*)carve((size_t)NE * NB * 4);
  int*    win  = (int*)carve((size_t)NB * 4);

  const int T = 256;
  // Pre-pass conversions
  k_cvt_f32_bf16<<<dim3(16384), T, 0, stream>>>(x, xbf, (size_t)NB * ND);
  k_cvt_f32_bf16<<<dim3(8192), T, 0, stream>>>(W_align, Wa, (size_t)NE * ND * ND);
  k_cvt_f32_bf16<<<dim3(4096), T, 0, stream>>>(W_out, WoT, (size_t)NE * ND * RD);
  k_cvt_alignT<<<dim3(8192), T, 0, stream>>>(W_align, WaT);
  k_build_wcat<<<dim3(4096), T, 0, stream>>>(W_in, W_dt, W_phase, Wcat);
  k_zero_f32<<<dim3((NE * NB) / T), T, 0, stream>>>(err, (size_t)NE * NB);
  // Composite weight GEMM: Mt[e] = Wcat[e] (512xD) x W_align[e] (as B), K=D
  k_gemm_comp<<<dim3(64, NE), T, 0, stream>>>(Wcat, Wa, Mt);
  // K1: alignment error norms, 64-row blocks (MT=4 -> ~32 FLOP/B of L2 traffic)
  k_err<<<dim3(NB / 64, NE), T, 0, stream>>>(xbf, WaT, x, err);
  // K2: winner select
  k_winner<<<dim3(NB / T), T, 0, stream>>>(err, win);
  // K3: fused proj GEMM + recurrent update + winner masking (64 KB dynamic LDS)
  k_cat<<<dim3(NB / 32, NE), T, 32 * NCAT * sizeof(float), stream>>>(
      xbf, Mt, A_log, A_imag, state_re, state_im, win, NRm);
  // K4: output GEMM accumulated over experts
  k_out<<<dim3((NB / 32) * 2), T, 0, stream>>>(NRm, WoT, out);
}
